// GraphModel_26268019982828
// MI455X (gfx1250) — matrix operand order repaired and checked by reading, not compiled, not run on hardware
//
#include <hip/hip_runtime.h>

typedef __attribute__((ext_vector_type(16))) __bf16 v16bf;
typedef __attribute__((ext_vector_type(8)))  float  v8f;

__device__ __forceinline__ unsigned int f2bf_bits(float f) {
  unsigned int u = __float_as_uint(f);
  u += 0x7FFFu + ((u >> 16) & 1u);          // round-to-nearest-even
  return u >> 16;
}

// ---------------- small elementwise / graph kernels ----------------

__global__ void __launch_bounds__(256) k_fill1(float* p, int n) {
  int i = blockIdx.x * 256 + threadIdx.x;
  if (i < n) p[i] = 1.0f;
}

__global__ void __launch_bounds__(256) k_deg_edges(float* deg, const long long* col, int e) {
  int i = blockIdx.x * 256 + threadIdx.x;
  if (i < e) unsafeAtomicAdd(&deg[(int)col[i]], 1.0f);
}

__global__ void __launch_bounds__(256) k_rsqrt_inplace(float* p, int n) {
  int i = blockIdx.x * 256 + threadIdx.x;
  if (i < n) p[i] = rsqrtf(p[i]);
}

// n4 = count/4; converts 4 floats -> 4 packed bf16 per thread
template <bool RELU>
__global__ void __launch_bounds__(256) k_cvt4(const float* __restrict__ x,
                                              unsigned short* __restrict__ y, int n4) {
  int i = blockIdx.x * 256 + threadIdx.x;
  if (i < n4) {
    float4 v = reinterpret_cast<const float4*>(x)[i];
    if (RELU) {
      v.x = fmaxf(v.x, 0.f); v.y = fmaxf(v.y, 0.f);
      v.z = fmaxf(v.z, 0.f); v.w = fmaxf(v.w, 0.f);
    }
    uint2 o;
    o.x = f2bf_bits(v.x) | (f2bf_bits(v.y) << 16);
    o.y = f2bf_bits(v.z) | (f2bf_bits(v.w) << 16);
    reinterpret_cast<uint2*>(y)[i] = o;
  }
}

// Transpose + convert: y[n*K + k] = bf16(x[k*N + n]).  K,N multiples of 32.
__global__ void __launch_bounds__(256) k_cvtT(const float* __restrict__ x,
                                              unsigned short* __restrict__ y,
                                              int K, int N) {
  __shared__ float tile[32][33];
  const int k0 = blockIdx.y * 32;
  const int n0 = blockIdx.x * 32;
  const int r  = threadIdx.x >> 3;         // 0..31
  const int c4 = (threadIdx.x & 7) * 4;    // 0,4,..,28
  float4 v = *reinterpret_cast<const float4*>(x + (size_t)(k0 + r) * N + n0 + c4);
  tile[r][c4 + 0] = v.x; tile[r][c4 + 1] = v.y;
  tile[r][c4 + 2] = v.z; tile[r][c4 + 3] = v.w;
  __syncthreads();
  uint2 o;
  o.x = f2bf_bits(tile[c4 + 0][r]) | (f2bf_bits(tile[c4 + 1][r]) << 16);
  o.y = f2bf_bits(tile[c4 + 2][r]) | (f2bf_bits(tile[c4 + 3][r]) << 16);
  *reinterpret_cast<uint2*>(y + (size_t)(n0 + r) * K + k0 + c4) = o;
}

// agg[i,:] = bias + dinv[i]^2 * h[i,:]   (self-loop contribution + bias)
__global__ void __launch_bounds__(256) k_agg_init(float* __restrict__ agg,
                                                  const float* __restrict__ h,
                                                  const float* __restrict__ bias,
                                                  const float* __restrict__ dinv,
                                                  int n, int dshift) {
  int i = blockIdx.x * 256 + threadIdx.x;
  int d = 1 << dshift;
  if (i < (n << dshift)) {
    int node = i >> dshift, dd = i & (d - 1);
    float s = dinv[node];
    agg[i] = bias[dd] + s * s * h[i];
  }
}

// agg[col,:] += dinv[row]*dinv[col] * h[row,:]   (one block per edge)
__global__ void __launch_bounds__(256) k_agg_edges(float* __restrict__ agg,
                                                   const float* __restrict__ h,
                                                   const float* __restrict__ dinv,
                                                   const long long* __restrict__ row,
                                                   const long long* __restrict__ col,
                                                   int d) {
  int e = blockIdx.x;
  int r = (int)row[e], c = (int)col[e];
  float s = dinv[r] * dinv[c];
  const float4* hs = reinterpret_cast<const float4*>(h + (size_t)r * d);
  float* ad = agg + (size_t)c * d;
  for (int i = threadIdx.x; i < (d >> 2); i += 256) {
    float4 v = hs[i];
    unsafeAtomicAdd(&ad[i * 4 + 0], s * v.x);
    unsafeAtomicAdd(&ad[i * 4 + 1], s * v.y);
    unsafeAtomicAdd(&ad[i * 4 + 2], s * v.z);
    unsafeAtomicAdd(&ad[i * 4 + 3], s * v.w);
  }
}

// ---------------- bf16 WMMA GEMM ----------------
// C(MxN, f32, row-major) = A(MxK bf16 row-major) @ B^T, B given N-major (NxK).
// Block: 256 thr = 8 waves (4 along M x 2 along N), block tile 128x128,
// wave tile 32x64 (2x4 WMMA frags), K step 32.
// Double-buffered LDS (one barrier per K step) + register-staged global loads.
// LDS rows padded to 40 ushorts: 16B-aligned chunks, conflict-free columns.

#define BM 128
#define BN 128
#define BK 32
#define LDP 40

union FragU { uint4 q[2]; v16bf v; };

__global__ void __launch_bounds__(256)
k_gemm_bf16(const unsigned short* __restrict__ A,
            const unsigned short* __restrict__ B,
            float* __restrict__ C, int Mdim, int Ndim, int Kdim) {
  __shared__ unsigned short sA[2][BM][LDP];  // [buf][m][k]
  __shared__ unsigned short sB[2][BN][LDP];  // [buf][n][k]

  const int t    = threadIdx.x;
  const int lane = t & 31;
  const int half = lane >> 4;
  const int lm   = lane & 15;
  const int w    = t >> 5;
  const int m0   = (w >> 1) * 32;   // 4 waves along M
  const int n0   = (w & 1) * 64;    // 2 waves along N
  const int blockM = blockIdx.y * BM;
  const int blockN = blockIdx.x * BN;

  // identical K-major loaders for A and B: 16 contiguous bf16 per thread
  const int ldr = t >> 1;            // 0..127
  const int ldc = (t & 1) * 16;      // 0 or 16
  int gm = blockM + ldr; if (gm >= Mdim) gm = Mdim - 1;
  int gn = blockN + ldr; if (gn >= Ndim) gn = Ndim - 1;
  const unsigned short* Arow = A + (size_t)gm * Kdim + ldc;
  const unsigned short* Brow = B + (size_t)gn * Kdim + ldc;

  v8f acc[2][4] = {};
  uint4 ra0, ra1, rb0, rb1;

  auto gload = [&](int kk) {
    const uint4* pa = reinterpret_cast<const uint4*>(Arow + kk);
    const uint4* pb = reinterpret_cast<const uint4*>(Brow + kk);
    ra0 = pa[0]; ra1 = pa[1];
    rb0 = pb[0]; rb1 = pb[1];
  };
  auto stage = [&](int p) {
    *reinterpret_cast<uint4*>(&sA[p][ldr][ldc])     = ra0;
    *reinterpret_cast<uint4*>(&sA[p][ldr][ldc + 8]) = ra1;
    *reinterpret_cast<uint4*>(&sB[p][ldr][ldc])     = rb0;
    *reinterpret_cast<uint4*>(&sB[p][ldr][ldc + 8]) = rb1;
  };

  gload(0);
  stage(0);
  int p = 0;
  for (int kk = 0; kk < Kdim; kk += BK) {
    __syncthreads();                      // buffer p fully staged for all waves
    const bool more = (kk + BK) < Kdim;
    if (more) gload(kk + BK);             // next tile: global -> regs (overlaps)

    FragU af[2], bfr[4];
#pragma unroll
    for (int f = 0; f < 2; ++f) {         // A frags: K chunks half*8 and 16+half*8
      const unsigned short* pa = &sA[p][m0 + f * 16 + lm][0];
      af[f].q[0] = *reinterpret_cast<const uint4*>(pa + half * 8);
      af[f].q[1] = *reinterpret_cast<const uint4*>(pa + 16 + half * 8);
    }
#pragma unroll
    for (int f = 0; f < 4; ++f) {         // B frags: K chunks half*8 and 16+half*8
      const unsigned short* pb = &sB[p][n0 + f * 16 + lm][0];
      bfr[f].q[0] = *reinterpret_cast<const uint4*>(pb + half * 8);
      bfr[f].q[1] = *reinterpret_cast<const uint4*>(pb + 16 + half * 8);
    }
#pragma unroll
    for (int fm = 0; fm < 2; ++fm)
#pragma unroll
      for (int fn = 0; fn < 4; ++fn)
        acc[fm][fn] = __builtin_amdgcn_wmma_f32_16x16x32_bf16(
            false, af[fm].v, false, bfr[fn].v, (short)0, acc[fm][fn], false, false);

    if (more) stage(p ^ 1);               // write other buffer; no barrier needed
    p ^= 1;
  }

  // epilogue: uniform fast path for full tiles
  const bool full = (blockM + BM <= Mdim) && (blockN + BN <= Ndim);
#pragma unroll
  for (int fm = 0; fm < 2; ++fm) {
#pragma unroll
    for (int fn = 0; fn < 4; ++fn) {
      const int colI = blockN + n0 + fn * 16 + lm;
      const int row0 = blockM + m0 + fm * 16 + half * 8;
      float* ptr = C + (size_t)row0 * Ndim + colI;
      if (full) {
#pragma unroll
        for (int v = 0; v < 8; ++v) { *ptr = acc[fm][fn][v]; ptr += Ndim; }
      } else {
        const bool colOK = colI < Ndim;
#pragma unroll
        for (int v = 0; v < 8; ++v) {
          if (colOK && (row0 + v) < Mdim) *ptr = acc[fm][fn][v];
          ptr += Ndim;
        }
      }
    }
  }
}

// ---------------- host launcher ----------------

extern "C" void kernel_launch(void* const* d_in, const int* in_sizes, int n_in,
                              void* d_out, int out_size, void* d_ws, size_t ws_size,
                              hipStream_t stream) {
  const float*     nodes = (const float*)d_in[0];
  const long long* ei    = (const long long*)d_in[1];
  const float*     img   = (const float*)d_in[2];
  const float*     W1    = (const float*)d_in[3];
  const float*     b1    = (const float*)d_in[4];
  const float*     W2    = (const float*)d_in[5];
  const float*     b2    = (const float*)d_in[6];
  const float*     W3    = (const float*)d_in[7];
  const float*     b3    = (const float*)d_in[8];
  (void)n_in; (void)out_size; (void)ws_size;

  const int Dm = 512, Hm = 2048;           // shifts: 9, 11
  const int N     = in_sizes[0] / Dm;      // 28535
  const int E     = in_sizes[1] / 2;       // 113000
  const int NPAIR = N - 360;               // skip N_ATTRS + N_OBJS rows
  const int MIMG  = in_sizes[2] / Dm;      // 64

  char* ws = (char*)d_ws;
  auto al = [](size_t x) { return (x + 255) & ~(size_t)255; };
  size_t off = 0;
  float* dinv = (float*)(ws + off);                     off = al(off + (size_t)N * 4);
  float* bufH = (float*)(ws + off);                     off = al(off + (size_t)N * Hm * 4);
  float* bufG = (float*)(ws + off);                     off = al(off + (size_t)N * Hm * 4);
  unsigned short* bufA16 = (unsigned short*)(ws + off); off = al(off + (size_t)N * Hm * 2);
  unsigned short* bufW16 = (unsigned short*)(ws + off); off = al(off + (size_t)Hm * Hm * 2);

  const long long* rowp = ei;
  const long long* colp = ei + E;

  dim3 thr(256);
  auto nb = [](long long n) { return dim3((unsigned)((n + 255) / 256)); };
  const int gy = (N + BM - 1) / BM;        // 223 M-blocks for node GEMMs

  // degrees -> dinv (in place)
  k_fill1<<<nb(N), thr, 0, stream>>>(dinv, N);
  k_deg_edges<<<nb(E), thr, 0, stream>>>(dinv, colp, E);
  k_rsqrt_inplace<<<nb(N), thr, 0, stream>>>(dinv, N);

  // ---- layer 1: h = relu(Agg(nodes @ W1) + b1)
  k_cvt4<false><<<nb((long long)N * Dm / 4), thr, 0, stream>>>(nodes, bufA16, N * Dm / 4);
  k_cvtT<<<dim3(Hm / 32, Dm / 32), thr, 0, stream>>>(W1, bufW16, Dm, Hm);  // -> (Hm x Dm)
  k_gemm_bf16<<<dim3(Hm / BN, gy), thr, 0, stream>>>(bufA16, bufW16, bufH, N, Hm, Dm);
  k_agg_init<<<nb((long long)N * Hm), thr, 0, stream>>>(bufG, bufH, b1, dinv, N, 11);
  k_agg_edges<<<dim3(E), thr, 0, stream>>>(bufG, bufH, dinv, rowp, colp, Hm);
  k_cvt4<true><<<nb((long long)N * Hm / 4), thr, 0, stream>>>(bufG, bufA16, N * Hm / 4);

  // ---- layer 2
  k_cvtT<<<dim3(Hm / 32, Hm / 32), thr, 0, stream>>>(W2, bufW16, Hm, Hm);  // -> (Hm x Hm)
  k_gemm_bf16<<<dim3(Hm / BN, gy), thr, 0, stream>>>(bufA16, bufW16, bufH, N, Hm, Hm);
  k_agg_init<<<nb((long long)N * Hm), thr, 0, stream>>>(bufG, bufH, b2, dinv, N, 11);
  k_agg_edges<<<dim3(E), thr, 0, stream>>>(bufG, bufH, dinv, rowp, colp, Hm);
  k_cvt4<true><<<nb((long long)N * Hm / 4), thr, 0, stream>>>(bufG, bufA16, N * Hm / 4);

  // ---- layer 3 (no relu)
  k_cvtT<<<dim3(Dm / 32, Hm / 32), thr, 0, stream>>>(W3, bufW16, Hm, Dm);  // -> (Dm x Hm)
  k_gemm_bf16<<<dim3(Dm / BN, gy), thr, 0, stream>>>(bufA16, bufW16, bufH, N, Dm, Hm);
  k_agg_init<<<nb((long long)N * Dm), thr, 0, stream>>>(bufG, bufH, b3, dinv, N, 9);
  k_agg_edges<<<dim3(E), thr, 0, stream>>>(bufG, bufH, dinv, rowp, colp, Dm);

  // ---- final: out = img @ pair_embed^T  (pair_embed = bufG rows 360..N, N-major NxK)
  unsigned short* p16   = bufA16;                        // reuse (L3 input dead)
  unsigned short* img16 = bufA16 + (size_t)NPAIR * Dm;
  k_cvt4<false><<<nb((long long)NPAIR * Dm / 4), thr, 0, stream>>>(
      bufG + (size_t)360 * Dm, p16, NPAIR * Dm / 4);
  k_cvt4<false><<<nb((long long)MIMG * Dm / 4), thr, 0, stream>>>(img, img16, MIMG * Dm / 4);
  k_gemm_bf16<<<dim3((NPAIR + BN - 1) / BN, (MIMG + BM - 1) / BM), thr, 0, stream>>>(
      img16, p16, (float*)d_out, MIMG, NPAIR, Dm);
}
